// Inner_88957362634756
// MI455X (gfx1250) — compile-verified, matrix operand order checked by reading, not run on hardware
//
#include <hip/hip_runtime.h>

// Soft decision tree forward, fused:
//   gates -> leaf path probs -> 64x batched GEMM (bf16 WMMA) -> softmax -> mixture
// Sizes from reference: N=2048 rows, K=1024 features, 63 inner nodes, 64 leaves, 1000 outputs.

#define NROWS  2048
#define KDIM   1024
#define NOUT   1000
#define NPAD   1024   // padded output columns (64 tiles of 16)
#define NLEAF  64
#define NINNER 63
#define TDEPTH 6

typedef __attribute__((ext_vector_type(16))) __bf16 v16bf;
typedef __attribute__((ext_vector_type(8)))  float  v8f;

struct Frag32B { uint4 lo; uint4 hi; };

// Load one wave32 WMMA 16-bit operand fragment (v16bf) for this lane.
// 16-bit A/B layout: lanes 0-15 hold K chunks {c0..c0+7, c0+16..c0+23} with c0=0,
// lanes 16-31 the same with c0=8. Two contiguous 16-byte loads per lane.
__device__ __forceinline__ v16bf load_frag(const unsigned short* p, int c0) {
  Frag32B f;
  f.lo = *(const uint4*)(p + c0);
  f.hi = *(const uint4*)(p + c0 + 16);
  return __builtin_bit_cast(v16bf, f);
}

__device__ __forceinline__ unsigned short f2bf(float x) {  // RNE fp32 -> bf16
  unsigned u = __float_as_uint(x);
  u += 0x7FFFu + ((u >> 16) & 1u);
  return (unsigned short)(u >> 16);
}

// Order-preserving float<->uint encoding for LDS atomic max.
__device__ __forceinline__ unsigned encf(float f) {
  unsigned u = __float_as_uint(f);
  return (u & 0x80000000u) ? ~u : (u | 0x80000000u);
}
__device__ __forceinline__ float decf(unsigned u) {
  unsigned b = (u & 0x80000000u) ? (u & 0x7FFFFFFFu) : ~u;
  return __uint_as_float(b);
}

// ---------------- pre-pack kernels ----------------

// Wt[l][n][k] (bf16, n padded to 1024 with zeros) from W_leaf[l][k][n] (fp32).
// Tiled 32x32 LDS transpose: coalesced reads along n, coalesced writes along k.
__global__ void pack_w_kernel(const float* __restrict__ W_leaf,
                              unsigned short* __restrict__ Wt) {
  __shared__ float tile[32][33];
  const int tx = threadIdx.x & 31;
  const int ty = threadIdx.x >> 5;          // 0..7
  const int nt = blockIdx.x;                // 32 n-tiles
  const int kt = blockIdx.y;                // 32 k-tiles
  const int l  = blockIdx.z;                // 64 leaves
#pragma unroll
  for (int r = 0; r < 4; ++r) {
    const int kk = ty + r * 8;
    const int k  = kt * 32 + kk;
    const int n  = nt * 32 + tx;
    tile[kk][tx] = (n < NOUT) ? W_leaf[((size_t)l * KDIM + k) * NOUT + n] : 0.f;
  }
  __syncthreads();
#pragma unroll
  for (int r = 0; r < 4; ++r) {
    const int nn = ty + r * 8;
    const int n  = nt * 32 + nn;
    const int k  = kt * 32 + tx;
    Wt[((size_t)l * NPAD + n) * KDIM + k] = f2bf(tile[tx][nn]);
  }
}

__global__ void pack_x_kernel(const float* __restrict__ x,
                              unsigned short* __restrict__ xbf) {
  const int idx = blockIdx.x * blockDim.x + threadIdx.x;
  if (idx < NROWS * KDIM) xbf[idx] = f2bf(x[idx]);
}

// ---------------- gating ----------------

// One wave per (row, inner node): 1024-long dot product + sigmoid.
__global__ void gates_kernel(const float* __restrict__ x,
                             const float* __restrict__ W_in,
                             const float* __restrict__ b_in,
                             const float* __restrict__ beta,
                             float* __restrict__ g) {
  const int wid  = (blockIdx.x * blockDim.x + threadIdx.x) >> 5;
  const int lane = threadIdx.x & 31;
  if (wid >= NROWS * NINNER) return;        // uniform per wave
  const int n = wid / NINNER;
  const int k = wid % NINNER;
  const float* xr = x + (size_t)n * KDIM;
  const float* wr = W_in + (size_t)k * KDIM;
  float s = 0.f;
  for (int t = lane; t < KDIM; t += 32) s += xr[t] * wr[t];
#pragma unroll
  for (int m = 16; m >= 1; m >>= 1) s += __shfl_xor(s, m, 32);
  if (lane == 0) {
    const float z = beta[k] * (s + b_in[k]);
    g[wid] = 1.f / (1.f + __expf(-z));
  }
}

// leaf_p[n][l] = product of gate probs along root->leaf path.
__global__ void leafp_kernel(const float* __restrict__ g,
                             float* __restrict__ leafp) {
  const int idx = blockIdx.x * blockDim.x + threadIdx.x;
  if (idx >= NROWS * NLEAF) return;
  const int n = idx >> 6;
  const int l = idx & (NLEAF - 1);
  const float* gn = g + (size_t)n * NINNER;
  int node = 0;
  float p = 1.f;
#pragma unroll
  for (int k = 0; k < TDEPTH; ++k) {
    const int b = (l >> (TDEPTH - 1 - k)) & 1;
    const float gv = gn[node];
    p *= b ? (1.f - gv) : gv;
    node = 2 * node + 1 + b;
  }
  leafp[idx] = p;
}

__global__ void zero_out_kernel(float* __restrict__ out, int n) {
  const int idx = blockIdx.x * blockDim.x + threadIdx.x;
  if (idx < n) out[idx] = 0.f;
}

// ---------------- fused main kernel ----------------
// Grid: (64 M-tiles of 32 rows, 8 leaf-groups of 8 leaves). Block: 512 (16 wave32).
// Each wave owns 4 column tiles (64 cols); 16 waves cover 1024 padded cols.
// Per leaf: WMMA GEMM over K=1024, add bias, cross-wave softmax via LDS,
// accumulate leaf_p-weighted distribution into registers; one atomicAdd per
// output element per leaf-group at the end.
__global__ __launch_bounds__(512) void soft_tree_main(
    const unsigned short* __restrict__ Wt,    // [64][1024][1024] bf16
    const unsigned short* __restrict__ xbf,   // [2048][1024] bf16
    const float* __restrict__ b_leaf,         // [64][1000]
    const float* __restrict__ leafp,          // [2048][64]
    float* __restrict__ out) {                // [2048][1000]
  __shared__ unsigned smax[32];
  __shared__ float    ssum[32];
  __shared__ float    srat[32];

  const int tid    = threadIdx.x;
  const int wave   = tid >> 5;
  const int lane   = tid & 31;
  const int half   = lane >> 4;     // which 16-lane half
  const int ln     = lane & 15;     // M index (A) / N index (B,C)
  const int c0     = half * 8;      // K-chunk base for 16-bit fragments
  const int m_base = blockIdx.x * 32;
  const int lbase  = blockIdx.y * 8;

  v8f outacc[2][4];
#pragma unroll
  for (int rt = 0; rt < 2; ++rt)
#pragma unroll
    for (int ct = 0; ct < 4; ++ct)
#pragma unroll
      for (int j = 0; j < 8; ++j) outacc[rt][ct][j] = 0.f;

  for (int lg = 0; lg < 8; ++lg) {
    const int l = lbase + lg;
    const unsigned short* Wl = Wt + (size_t)l * NPAD * KDIM;

    v8f acc[2][4];
#pragma unroll
    for (int rt = 0; rt < 2; ++rt)
#pragma unroll
      for (int ct = 0; ct < 4; ++ct)
#pragma unroll
        for (int j = 0; j < 8; ++j) acc[rt][ct][j] = 0.f;

    // ---- GEMM: logits[32 x 64cols] += x[32 x 1024] * W_l[1024 x 64cols] ----
    for (int ks = 0; ks < KDIM / 32; ++ks) {
      const int kb = ks * 32;
      v16bf a[2];
#pragma unroll
      for (int rt = 0; rt < 2; ++rt)
        a[rt] = load_frag(xbf + (size_t)(m_base + rt * 16 + ln) * KDIM + kb, c0);
#pragma unroll
      for (int ct = 0; ct < 4; ++ct) {
        const int n = (wave * 4 + ct) * 16 + ln;
        const v16bf b = load_frag(Wl + (size_t)n * KDIM + kb, c0);
        acc[0][ct] = __builtin_amdgcn_wmma_f32_16x16x32_bf16(
            false, a[0], false, b, (short)0, acc[0][ct], false, false);
        acc[1][ct] = __builtin_amdgcn_wmma_f32_16x16x32_bf16(
            false, a[1], false, b, (short)0, acc[1][ct], false, false);
      }
    }

    // ---- bias (padded cols get -1e30 so exp -> 0) ----
    float bn[4];
#pragma unroll
    for (int ct = 0; ct < 4; ++ct) {
      const int n = (wave * 4 + ct) * 16 + ln;
      bn[ct] = (n < NOUT) ? b_leaf[(size_t)l * NOUT + n] : -1e30f;
    }
#pragma unroll
    for (int rt = 0; rt < 2; ++rt)
#pragma unroll
      for (int ct = 0; ct < 4; ++ct)
#pragma unroll
        for (int j = 0; j < 8; ++j) acc[rt][ct][j] += bn[ct];

    if (tid < 32) { smax[tid] = 0u; ssum[tid] = 0.f; }
    __syncthreads();

    // ---- row max: registers -> shfl over 16-lane half -> 2 LDS atomics/wave ----
#pragma unroll
    for (int rt = 0; rt < 2; ++rt) {
#pragma unroll
      for (int j = 0; j < 8; ++j) {
        float m = acc[rt][0][j];
        m = fmaxf(m, acc[rt][1][j]);
        m = fmaxf(m, acc[rt][2][j]);
        m = fmaxf(m, acc[rt][3][j]);
#pragma unroll
        for (int s = 1; s < 16; s <<= 1) m = fmaxf(m, __shfl_xor(m, s, 32));
        if (ln == 0) atomicMax(&smax[rt * 16 + j + half * 8], encf(m));
      }
    }
    __syncthreads();

    // ---- exp + row sum ----
#pragma unroll
    for (int rt = 0; rt < 2; ++rt) {
#pragma unroll
      for (int j = 0; j < 8; ++j) {
        const float mv = decf(smax[rt * 16 + j + half * 8]);
        float s = 0.f;
#pragma unroll
        for (int ct = 0; ct < 4; ++ct) {
          const float e = __expf(acc[rt][ct][j] - mv);
          acc[rt][ct][j] = e;
          s += e;
        }
#pragma unroll
        for (int sh = 1; sh < 16; sh <<= 1) s += __shfl_xor(s, sh, 32);
        if (ln == 0) atomicAdd(&ssum[rt * 16 + j + half * 8], s);
      }
    }
    __syncthreads();

    if (tid < 32)
      srat[tid] = leafp[(size_t)(m_base + tid) * NLEAF + l] / ssum[tid];
    __syncthreads();

    // ---- out += leaf_p * softmax ----
#pragma unroll
    for (int rt = 0; rt < 2; ++rt) {
#pragma unroll
      for (int j = 0; j < 8; ++j) {
        const float r = srat[rt * 16 + j + half * 8];
#pragma unroll
        for (int ct = 0; ct < 4; ++ct) outacc[rt][ct][j] += acc[rt][ct][j] * r;
      }
    }
    __syncthreads();   // protect LDS reuse next leaf
  }

  // ---- one global atomic add per element per leaf-group ----
#pragma unroll
  for (int rt = 0; rt < 2; ++rt) {
#pragma unroll
    for (int ct = 0; ct < 4; ++ct) {
      const int n = (wave * 4 + ct) * 16 + ln;
      if (n < NOUT) {
#pragma unroll
        for (int j = 0; j < 8; ++j) {
          const int row = m_base + rt * 16 + j + half * 8;
          atomicAdd(&out[(size_t)row * NOUT + n], outacc[rt][ct][j]);
        }
      }
    }
  }
}

// ---------------- launch ----------------

extern "C" void kernel_launch(void* const* d_in, const int* in_sizes, int n_in,
                              void* d_out, int out_size, void* d_ws, size_t ws_size,
                              hipStream_t stream) {
  const float* x      = (const float*)d_in[0];
  const float* W_in   = (const float*)d_in[1];
  const float* b_in   = (const float*)d_in[2];
  const float* beta   = (const float*)d_in[3];
  const float* W_leaf = (const float*)d_in[4];
  const float* b_leaf = (const float*)d_in[5];
  float* out = (float*)d_out;

  char* ws = (char*)d_ws;
  const size_t WT_BYTES  = (size_t)NLEAF * NPAD * KDIM * 2;   // 134 MB bf16 (fits MI455X 192MB L2)
  const size_t XBF_BYTES = (size_t)NROWS * KDIM * 2;          // 4 MB
  const size_t G_BYTES   = (size_t)NROWS * NINNER * 4;
  unsigned short* Wt  = (unsigned short*)ws;
  unsigned short* xbf = (unsigned short*)(ws + WT_BYTES);
  float* g  = (float*)(ws + WT_BYTES + XBF_BYTES);
  float* lp = (float*)(ws + WT_BYTES + XBF_BYTES + G_BYTES);

  // 1) pack W_leaf -> bf16 transposed [l][n][k], padded to 1024 cols
  pack_w_kernel<<<dim3(32, 32, NLEAF), 256, 0, stream>>>(W_leaf, Wt);
  // 2) pack x -> bf16
  pack_x_kernel<<<(NROWS * KDIM + 255) / 256, 256, 0, stream>>>(x, xbf);
  // 3) gates (one wave per (row, node))
  {
    const int waves = NROWS * NINNER;
    gates_kernel<<<(waves * 32 + 255) / 256, 256, 0, stream>>>(x, W_in, b_in, beta, g);
  }
  // 4) leaf path probabilities
  leafp_kernel<<<(NROWS * NLEAF + 255) / 256, 256, 0, stream>>>(g, lp);
  // 5) zero output (harness poisons it)
  zero_out_kernel<<<(NROWS * NOUT + 255) / 256, 256, 0, stream>>>(out, NROWS * NOUT);
  // 6) fused WMMA GEMM + softmax + mixture
  soft_tree_main<<<dim3(NROWS / 32, NLEAF / 8), 512, 0, stream>>>(Wt, xbf, b_leaf, lp, out);
}
